// MPNNs_85143431676130
// MI455X (gfx1250) — compile-verified, matrix operand order checked by reading, not run on hardware
//
#include <hip/hip_runtime.h>

// ---------------------------------------------------------------------------
// MPNN MetaLayer step on gfx1250 using bf16 WMMA (V_WMMA_F32_16X16X32_BF16).
// Weights pre-swizzled into B-fragment layout (one 32B load per lane/frag);
// edge kernel uses M=32 per wave so each B fragment feeds two WMMAs.
// ---------------------------------------------------------------------------

typedef __attribute__((ext_vector_type(16))) __bf16 v16bf;
typedef __attribute__((ext_vector_type(8)))  __bf16 v8bf;
typedef __attribute__((ext_vector_type(8)))  float  v8f;

#define N_NODES  50000
#define N_EDGES  800000
#define N_GRAPHS 256

// ---- fragment helpers (ISA 7.12.2 layouts, wave32) ------------------------
// 16-bit A 16x32: lane<16 holds row M=lane, K = {0..7,16..23}; lane+16 holds
// K = {8..15,24..31}.  Element i of the v16bf maps to:
//   i in [0,8):  k = half*8 + i
//   i in [8,16): k = 16 + half*8 + (i-8)
__device__ __forceinline__ int frag_k(int i, int half) {
  return ((i >= 8) ? 16 : 0) + (((i >> 1) & 3) << 1) + (i & 1) + half * 8;
}

__device__ __forceinline__ v8f zero8() {
  v8f z = {0.f, 0.f, 0.f, 0.f, 0.f, 0.f, 0.f, 0.f};
  return z;
}

// B fragment from fragment-packed weights: fragment (kt,nt) stores lane l's
// 16 bf16 values contiguously at ((nt*nkt + kt)*32 + l)*16 elements.
__device__ __forceinline__ v16bf load_b_frag(const __bf16* __restrict__ w,
                                             int nkt, int kt, int nt,
                                             int lane) {
  const v16bf* p = (const v16bf*)w;
  return p[((nt * nkt + kt) << 5) + lane];
}

// A fragment from a row-major [rows x ld] bf16 tile in LDS starting at row
// rbase: two contiguous 8-element (16-byte) runs -> ds_load_b128 x2.
__device__ __forceinline__ v16bf load_a_lds(const __bf16* s, int ld, int k0,
                                            int rbase, int lane) {
  const int m = rbase + (lane & 15);
  const int half = lane >> 4;
  const __bf16* base = s + m * ld + k0 + half * 8;
  v8bf lo = *(const v8bf*)base;
  v8bf hi = *(const v8bf*)(base + 16);
  v16bf r;
#pragma unroll
  for (int j = 0; j < 8; ++j) {
    r[j] = lo[j];
    r[j + 8] = hi[j];
  }
  return r;
}

__device__ __forceinline__ v8f wmma_bf16(v16bf a, v16bf b, v8f c) {
  return __builtin_amdgcn_wmma_f32_16x16x32_bf16(
      /*neg_a=*/false, a, /*neg_b=*/false, b,
      /*c_mod=*/(short)0, c, /*reuse_a=*/false, /*reuse_b=*/false);
}

// Load one 8-float run (two float4) and convert into afrag elements
// [rr*8, rr*8+8) with a scale factor; p==nullptr means zero padding.
__device__ __forceinline__ void run_to_frag(v16bf& frag, int rr,
                                            const float* p, float scale) {
  if (p) {
    float4 lo = *(const float4*)p;
    float4 hi = *(const float4*)(p + 4);
    frag[rr * 8 + 0] = (__bf16)(lo.x * scale);
    frag[rr * 8 + 1] = (__bf16)(lo.y * scale);
    frag[rr * 8 + 2] = (__bf16)(lo.z * scale);
    frag[rr * 8 + 3] = (__bf16)(lo.w * scale);
    frag[rr * 8 + 4] = (__bf16)(hi.x * scale);
    frag[rr * 8 + 5] = (__bf16)(hi.y * scale);
    frag[rr * 8 + 6] = (__bf16)(hi.z * scale);
    frag[rr * 8 + 7] = (__bf16)(hi.w * scale);
  } else {
#pragma unroll
    for (int j = 0; j < 8; ++j) frag[rr * 8 + j] = (__bf16)0.f;
  }
}

// ---------------------------------------------------------------------------
// Setup kernels.
// ---------------------------------------------------------------------------
// Convert [K x N] f32 weight into fragment-packed bf16 (K zero-padded to Kpad).
__global__ void cvt_w_frag(const float* __restrict__ src,
                           __bf16* __restrict__ dst, int K, int N, int Kpad) {
  int t = blockIdx.x * blockDim.x + threadIdx.x;
  if (t >= Kpad * N) return;
  int i = t & 15;
  int lane = (t >> 4) & 31;
  int f = t >> 9;                     // 32 lanes * 16 elems = 512 per fragment
  int nkt = Kpad >> 5;
  int nt = f / nkt;
  int kt = f - nt * nkt;
  int k = kt * 32 + frag_k(i, lane >> 4);
  int n = nt * 16 + (lane & 15);
  dst[t] = (k < K) ? (__bf16)src[k * N + n] : (__bf16)0.f;
}

__global__ void zero_k(float* __restrict__ p, int n) {
  int i = blockIdx.x * blockDim.x + threadIdx.x;
  if (i < n) p[i] = 0.f;
}

// ---------------------------------------------------------------------------
// Kernel 1: per 32-edge tile (one wave, two 16-row A blocks sharing B):
//   edge_new = relu([x[r],x[c],ea,u[b[r]]] @ eW1 + eb1) @ eW2 + eb2
//   msg      = relu([x[r],edge_new] @ nW1 + nb1) @ nW2 + nb2
//   agg[col] += msg (atomic), cnt[col] += 1
// ---------------------------------------------------------------------------
__global__ void __launch_bounds__(128) edge_kernel(
    const float* __restrict__ x, const float* __restrict__ edge_attr,
    const float* __restrict__ u,
    const __bf16* __restrict__ ew1, const float* __restrict__ eb1,
    const __bf16* __restrict__ ew2, const float* __restrict__ eb2,
    const __bf16* __restrict__ nw1, const float* __restrict__ nb1,
    const __bf16* __restrict__ nw2, const float* __restrict__ nb2,
    const int* __restrict__ row_idx, const int* __restrict__ col_idx,
    const int* __restrict__ batch,
    float* __restrict__ edge_out, float* __restrict__ agg,
    float* __restrict__ cnt) {
  __shared__ __bf16 smem[4][32 * 128 + 32 * 64];  // per-wave h[32x128], e[32x64]
  const int wave = threadIdx.x >> 5;
  const int lane = threadIdx.x & 31;
  const int tile = blockIdx.x * 4 + wave;
  if (tile >= N_EDGES / 32) return;
  __bf16* h_s = smem[wave];             // [32 x 128]
  __bf16* e_s = smem[wave] + 32 * 128;  // [32 x 64]

  // Warm the WGP cache with the weight blocks (global_prefetch_b8 path).
  __builtin_prefetch(ew1 + (size_t)lane * 64, 0, 1);
  __builtin_prefetch(ew2 + (size_t)lane * 64, 0, 1);
  __builtin_prefetch(nw1 + (size_t)lane * 64, 0, 1);
  __builtin_prefetch(nw2 + (size_t)lane * 64, 0, 1);

  const int m = lane & 15;
  const int half = lane >> 4;
  const int mb = half * 8;

  // Gather concat inputs into A fragments for both row-blocks
  // (K padded 176 -> 192; boundaries 64/128/160/176 are multiples of 8).
  int c_rb[2];
  v16bf afrag[2][6];
#pragma unroll
  for (int rb = 0; rb < 2; ++rb) {
    const int e = tile * 32 + rb * 16 + m;
    const int r = row_idx[e];
    c_rb[rb] = col_idx[e];
    const float* xr = x + (size_t)r * 64;
    const float* xc = x + (size_t)c_rb[rb] * 64;
    const float* ea = edge_attr + (size_t)e * 32;
    const float* ub = u + batch[r] * 16;
#pragma unroll
    for (int kt = 0; kt < 6; ++kt) {
#pragma unroll
      for (int rr = 0; rr < 2; ++rr) {
        const int k0 = kt * 32 + rr * 16 + half * 8;
        const float* p;
        if (k0 < 64)       p = xr + k0;
        else if (k0 < 128) p = xc + (k0 - 64);
        else if (k0 < 160) p = ea + (k0 - 128);
        else if (k0 < 176) p = ub + (k0 - 160);
        else               p = nullptr;
        run_to_frag(afrag[rb][kt], rr, p, 1.f);
      }
    }
  }

  // Edge MLP layer 1: 176 -> 128, ReLU; each B fragment feeds 2 WMMAs.
#pragma unroll
  for (int nt = 0; nt < 8; ++nt) {
    v8f acc0 = zero8(), acc1 = zero8();
#pragma unroll
    for (int kt = 0; kt < 6; ++kt) {
      v16bf b = load_b_frag(ew1, 6, kt, nt, lane);
      acc0 = wmma_bf16(afrag[0][kt], b, acc0);
      acc1 = wmma_bf16(afrag[1][kt], b, acc1);
    }
    const int n = nt * 16 + m;
    const float bias = eb1[n];
#pragma unroll
    for (int q = 0; q < 8; ++q) {
      float v0 = acc0[q] + bias;
      float v1 = acc1[q] + bias;
      h_s[(mb + q) * 128 + n]        = (__bf16)(v0 > 0.f ? v0 : 0.f);
      h_s[(16 + mb + q) * 128 + n]   = (__bf16)(v1 > 0.f ? v1 : 0.f);
    }
  }
  asm volatile("s_wait_dscnt 0" ::: "memory");

  // Edge MLP layer 2: 128 -> 64; write edge_new and keep bf16 copy in LDS.
  {
    v16bf ha[2][4];
#pragma unroll
    for (int rb = 0; rb < 2; ++rb)
#pragma unroll
      for (int kt = 0; kt < 4; ++kt)
        ha[rb][kt] = load_a_lds(h_s, 128, kt * 32, rb * 16, lane);
#pragma unroll
    for (int nt = 0; nt < 4; ++nt) {
      v8f acc0 = zero8(), acc1 = zero8();
#pragma unroll
      for (int kt = 0; kt < 4; ++kt) {
        v16bf b = load_b_frag(ew2, 4, kt, nt, lane);
        acc0 = wmma_bf16(ha[0][kt], b, acc0);
        acc1 = wmma_bf16(ha[1][kt], b, acc1);
      }
      const int n = nt * 16 + m;
      const float bias = eb2[n];
#pragma unroll
      for (int q = 0; q < 8; ++q) {
        float v0 = acc0[q] + bias;
        float v1 = acc1[q] + bias;
        edge_out[(size_t)(tile * 32 + mb + q) * 64 + n] = v0;
        edge_out[(size_t)(tile * 32 + 16 + mb + q) * 64 + n] = v1;
        e_s[(mb + q) * 64 + n] = (__bf16)v0;
        e_s[(16 + mb + q) * 64 + n] = (__bf16)v1;
      }
    }
  }
  asm volatile("s_wait_dscnt 0" ::: "memory");

  // Node-message MLP layer 1: [x[row], edge_new] (128) -> 128, ReLU.
  {
    v16bf ma[2][4];
#pragma unroll
    for (int rb = 0; rb < 2; ++rb) {
      ma[rb][0] = afrag[rb][0];
      ma[rb][1] = afrag[rb][1];
      ma[rb][2] = load_a_lds(e_s, 64, 0, rb * 16, lane);
      ma[rb][3] = load_a_lds(e_s, 64, 32, rb * 16, lane);
    }
#pragma unroll
    for (int nt = 0; nt < 8; ++nt) {
      v8f acc0 = zero8(), acc1 = zero8();
#pragma unroll
      for (int kt = 0; kt < 4; ++kt) {
        v16bf b = load_b_frag(nw1, 4, kt, nt, lane);
        acc0 = wmma_bf16(ma[0][kt], b, acc0);
        acc1 = wmma_bf16(ma[1][kt], b, acc1);
      }
      const int n = nt * 16 + m;
      const float bias = nb1[n];
#pragma unroll
      for (int q = 0; q < 8; ++q) {
        float v0 = acc0[q] + bias;
        float v1 = acc1[q] + bias;
        h_s[(mb + q) * 128 + n]      = (__bf16)(v0 > 0.f ? v0 : 0.f);
        h_s[(16 + mb + q) * 128 + n] = (__bf16)(v1 > 0.f ? v1 : 0.f);
      }
    }
  }
  asm volatile("s_wait_dscnt 0" ::: "memory");

  // Node-message MLP layer 2: 128 -> 128, scatter-add into agg[col].
  {
    v16bf ha[2][4];
#pragma unroll
    for (int rb = 0; rb < 2; ++rb)
#pragma unroll
      for (int kt = 0; kt < 4; ++kt)
        ha[rb][kt] = load_a_lds(h_s, 128, kt * 32, rb * 16, lane);
    int dq[2][8];
#pragma unroll
    for (int rb = 0; rb < 2; ++rb)
#pragma unroll
      for (int q = 0; q < 8; ++q)
        dq[rb][q] = col_idx[tile * 32 + rb * 16 + mb + q];
#pragma unroll
    for (int nt = 0; nt < 8; ++nt) {
      v8f acc0 = zero8(), acc1 = zero8();
#pragma unroll
      for (int kt = 0; kt < 4; ++kt) {
        v16bf b = load_b_frag(nw2, 4, kt, nt, lane);
        acc0 = wmma_bf16(ha[0][kt], b, acc0);
        acc1 = wmma_bf16(ha[1][kt], b, acc1);
      }
      const int n = nt * 16 + m;
      const float bias = nb2[n];
#pragma unroll
      for (int q = 0; q < 8; ++q) {
        atomicAdd(&agg[(size_t)dq[0][q] * 128 + n], acc0[q] + bias);
        atomicAdd(&agg[(size_t)dq[1][q] * 128 + n], acc1[q] + bias);
      }
    }
  }
  if (lane < 16) {
    atomicAdd(&cnt[c_rb[0]], 1.0f);
    atomicAdd(&cnt[c_rb[1]], 1.0f);
  }
}

// ---------------------------------------------------------------------------
// Kernel 2: per 16-node tile:
//   x_new = relu([x, agg/max(cnt,1), u[batch]] @ n2W1 + b1) @ n2W2 + b2
//   gsum[batch] += x_new, gcnt[batch] += 1
// ---------------------------------------------------------------------------
__global__ void __launch_bounds__(128) node_kernel(
    const float* __restrict__ x, const float* __restrict__ u,
    const float* __restrict__ agg, const float* __restrict__ cnt,
    const __bf16* __restrict__ w1, const float* __restrict__ b1,
    const __bf16* __restrict__ w2, const float* __restrict__ b2,
    const int* __restrict__ batch, float* __restrict__ x_out,
    float* __restrict__ gsum, float* __restrict__ gcnt) {
  __shared__ __bf16 smem[4][16 * 128];
  const int wave = threadIdx.x >> 5;
  const int lane = threadIdx.x & 31;
  const int tile = blockIdx.x * 4 + wave;
  if (tile >= N_NODES / 16) return;
  __bf16* h_s = smem[wave];

  const int m = lane & 15;
  const int half = lane >> 4;
  const int node = tile * 16 + m;
  const float* xr = x + (size_t)node * 64;
  const float* ag = agg + (size_t)node * 128;
  const float inv = 1.f / fmaxf(cnt[node], 1.f);
  const float* ub = u + batch[node] * 16;

  // Concat input 208 wide, padded to 224 (7 K-tiles). Boundaries 64/192/208
  // are multiples of 8 -> float4 run loads, with agg scaled by 1/cnt.
  v16bf afrag[7];
#pragma unroll
  for (int kt = 0; kt < 7; ++kt) {
#pragma unroll
    for (int rr = 0; rr < 2; ++rr) {
      const int k0 = kt * 32 + rr * 16 + half * 8;
      const float* p;
      float s = 1.f;
      if (k0 < 64)        p = xr + k0;
      else if (k0 < 192) { p = ag + (k0 - 64); s = inv; }
      else if (k0 < 208)  p = ub + (k0 - 192);
      else                p = nullptr;
      run_to_frag(afrag[kt], rr, p, s);
    }
  }

  // Layer 1: 208 -> 128, ReLU.
#pragma unroll
  for (int nt = 0; nt < 8; ++nt) {
    v8f acc = zero8();
#pragma unroll
    for (int kt = 0; kt < 7; ++kt)
      acc = wmma_bf16(afrag[kt], load_b_frag(w1, 7, kt, nt, lane), acc);
    const int n = nt * 16 + m;
    const float bias = b1[n];
    const int mb = half * 8;
#pragma unroll
    for (int q = 0; q < 8; ++q) {
      float v = acc[q] + bias;
      h_s[(mb + q) * 128 + n] = (__bf16)(v > 0.f ? v : 0.f);
    }
  }
  asm volatile("s_wait_dscnt 0" ::: "memory");

  // Layer 2: 128 -> 64; write x_new and pool per graph.
  v16bf ha[4];
#pragma unroll
  for (int kt = 0; kt < 4; ++kt)
    ha[kt] = load_a_lds(h_s, 128, kt * 32, 0, lane);
  const int mb = half * 8;
  int gq[8];
#pragma unroll
  for (int q = 0; q < 8; ++q) gq[q] = batch[tile * 16 + mb + q];
#pragma unroll
  for (int nt = 0; nt < 4; ++nt) {
    v8f acc = zero8();
#pragma unroll
    for (int kt = 0; kt < 4; ++kt)
      acc = wmma_bf16(ha[kt], load_b_frag(w2, 4, kt, nt, lane), acc);
    const int n = nt * 16 + m;
    const float bias = b2[n];
#pragma unroll
    for (int q = 0; q < 8; ++q) {
      float v = acc[q] + bias;
      x_out[(size_t)(tile * 16 + mb + q) * 64 + n] = v;
      atomicAdd(&gsum[gq[q] * 64 + n], v);
    }
  }
  if (lane < 16) atomicAdd(&gcnt[batch[node]], 1.f);
}

// ---------------------------------------------------------------------------
// Kernel 3: per 16-graph tile:
//   u_new = relu([u, gsum/max(gcnt,1)] @ gW1 + b1) @ gW2 + b2
// ---------------------------------------------------------------------------
__global__ void __launch_bounds__(128) global_kernel(
    const float* __restrict__ u, const float* __restrict__ gsum,
    const float* __restrict__ gcnt, const __bf16* __restrict__ w1,
    const float* __restrict__ b1, const __bf16* __restrict__ w2,
    const float* __restrict__ b2, float* __restrict__ u_out) {
  __shared__ __bf16 smem[4][16 * 128];
  const int wave = threadIdx.x >> 5;
  const int lane = threadIdx.x & 31;
  const int tile = blockIdx.x * 4 + wave;
  if (tile >= N_GRAPHS / 16) return;
  __bf16* h_s = smem[wave];

  const int m = lane & 15;
  const int half = lane >> 4;
  const int g = tile * 16 + m;
  const float inv = 1.f / fmaxf(gcnt[g], 1.f);
  const float* ug = u + g * 16;
  const float* gs = gsum + g * 64;

  // Concat input 80 wide, padded to 96 (3 K-tiles). Boundaries 16/80 are
  // multiples of 8.
  v16bf afrag[3];
#pragma unroll
  for (int kt = 0; kt < 3; ++kt) {
#pragma unroll
    for (int rr = 0; rr < 2; ++rr) {
      const int k0 = kt * 32 + rr * 16 + half * 8;
      const float* p;
      float s = 1.f;
      if (k0 < 16)       p = ug + k0;
      else if (k0 < 80) { p = gs + (k0 - 16); s = inv; }
      else               p = nullptr;
      run_to_frag(afrag[kt], rr, p, s);
    }
  }

#pragma unroll
  for (int nt = 0; nt < 8; ++nt) {
    v8f acc = zero8();
#pragma unroll
    for (int kt = 0; kt < 3; ++kt)
      acc = wmma_bf16(afrag[kt], load_b_frag(w1, 3, kt, nt, lane), acc);
    const int n = nt * 16 + m;
    const float bias = b1[n];
    const int mb = half * 8;
#pragma unroll
    for (int q = 0; q < 8; ++q) {
      float v = acc[q] + bias;
      h_s[(mb + q) * 128 + n] = (__bf16)(v > 0.f ? v : 0.f);
    }
  }
  asm volatile("s_wait_dscnt 0" ::: "memory");

  v16bf ha[4];
#pragma unroll
  for (int kt = 0; kt < 4; ++kt)
    ha[kt] = load_a_lds(h_s, 128, kt * 32, 0, lane);
#pragma unroll
  for (int nt = 0; nt < 2; ++nt) {
    v8f acc = zero8();
#pragma unroll
    for (int kt = 0; kt < 4; ++kt)
      acc = wmma_bf16(ha[kt], load_b_frag(w2, 4, kt, nt, lane), acc);
    const int n = nt * 16 + m;
    const float bias = b2[n];
    const int mb = half * 8;
#pragma unroll
    for (int q = 0; q < 8; ++q)
      u_out[(tile * 16 + mb + q) * 32 + n] = acc[q] + bias;
  }
}

// ---------------------------------------------------------------------------
// Host-side orchestration.
// ---------------------------------------------------------------------------
extern "C" void kernel_launch(void* const* d_in, const int* in_sizes, int n_in,
                              void* d_out, int out_size, void* d_ws,
                              size_t ws_size, hipStream_t stream) {
  (void)in_sizes; (void)n_in; (void)out_size; (void)ws_size;
  const float* x         = (const float*)d_in[0];
  const float* edge_attr = (const float*)d_in[1];
  const float* u         = (const float*)d_in[2];
  const float* e_w1  = (const float*)d_in[3];
  const float* e_b1  = (const float*)d_in[4];
  const float* e_w2  = (const float*)d_in[5];
  const float* e_b2  = (const float*)d_in[6];
  const float* n1_w1 = (const float*)d_in[7];
  const float* n1_b1 = (const float*)d_in[8];
  const float* n1_w2 = (const float*)d_in[9];
  const float* n1_b2 = (const float*)d_in[10];
  const float* n2_w1 = (const float*)d_in[11];
  const float* n2_b1 = (const float*)d_in[12];
  const float* n2_w2 = (const float*)d_in[13];
  const float* n2_b2 = (const float*)d_in[14];
  const float* g_w1  = (const float*)d_in[15];
  const float* g_b1  = (const float*)d_in[16];
  const float* g_w2  = (const float*)d_in[17];
  const float* g_b2  = (const float*)d_in[18];
  const int* ei      = (const int*)d_in[19];
  const int* batch   = (const int*)d_in[20];
  const int* row_idx = ei;
  const int* col_idx = ei + N_EDGES;

  float* out      = (float*)d_out;
  float* x_new    = out;                                     // [50000 x 64]
  float* edge_new = out + (size_t)N_NODES * 64;              // [800000 x 64]
  float* u_new    = out + (size_t)N_NODES * 64 + (size_t)N_EDGES * 64;  // [256 x 32]

  // Workspace layout (bytes; all weight blocks 32B-aligned).
  char* ws = (char*)d_ws;
  __bf16* ew1b  = (__bf16*)(ws + 0);        // 192x128 frag-packed
  __bf16* ew2b  = (__bf16*)(ws + 49152);    // 128x64
  __bf16* n1w1b = (__bf16*)(ws + 65536);    // 128x128
  __bf16* n1w2b = (__bf16*)(ws + 98304);    // 128x128
  __bf16* n2w1b = (__bf16*)(ws + 131072);   // 224x128
  __bf16* n2w2b = (__bf16*)(ws + 188416);   // 128x64
  __bf16* gw1b  = (__bf16*)(ws + 204800);   // 96x128
  __bf16* gw2b  = (__bf16*)(ws + 229376);   // 128x32
  float* agg  = (float*)(ws + 237568);      // [50000 x 128]
  float* cnt  = (float*)(ws + 237568 + 25600000);       // [50000]
  float* gsum = (float*)(ws + 237568 + 25800000);       // [256 x 64]
  float* gcnt = (float*)(ws + 237568 + 25865536);       // [256]

  // 1) Weight conversion into fragment-packed bf16.
  auto cvt = [&](const float* src, __bf16* dst, int K, int N, int Kpad) {
    int n = Kpad * N;
    cvt_w_frag<<<(n + 255) / 256, 256, 0, stream>>>(src, dst, K, N, Kpad);
  };
  cvt(e_w1, ew1b, 176, 128, 192);
  cvt(e_w2, ew2b, 128, 64, 128);
  cvt(n1_w1, n1w1b, 128, 128, 128);
  cvt(n1_w2, n1w2b, 128, 128, 128);
  cvt(n2_w1, n2w1b, 208, 128, 224);
  cvt(n2_w2, n2w2b, 128, 64, 128);
  cvt(g_w1, gw1b, 80, 128, 96);
  cvt(g_w2, gw2b, 128, 32, 128);

  // 2) Zero the atomic accumulators (agg, cnt, gsum, gcnt contiguous).
  {
    int nz = N_NODES * 128 + N_NODES + N_GRAPHS * 64 + N_GRAPHS;  // 6,466,640
    zero_k<<<(nz + 255) / 256, 256, 0, stream>>>(agg, nz);
  }

  // 3) Edge + message MLPs (25000 tiles of 32 edges, 4 waves/block).
  edge_kernel<<<N_EDGES / 32 / 4, 128, 0, stream>>>(
      x, edge_attr, u, ew1b, e_b1, ew2b, e_b2, n1w1b, n1_b1, n1w2b, n1_b2,
      row_idx, col_idx, batch, edge_new, agg, cnt);

  // 4) Node update (3125 tiles).
  node_kernel<<<(N_NODES / 16 + 3) / 4, 128, 0, stream>>>(
      x, u, agg, cnt, n2w1b, n2_b1, n2w2b, n2_b2, batch, x_new, gsum, gcnt);

  // 5) Global update (16 tiles).
  global_kernel<<<N_GRAPHS / 16 / 4, 128, 0, stream>>>(
      u, gsum, gcnt, gw1b, g_b1, gw2b, g_b2, u_new);
}
